// GemNetOCForceHead_704374636664
// MI455X (gfx1250) — matrix operand order repaired and checked by reading, not compiled, not run on hardware
//
#include <hip/hip_runtime.h>
#include <hip/hip_bf16.h>

// ---------------------------------------------------------------------------
// GemNet-OC force head, fused bf16-WMMA MLP + scatter kernels for gfx1250.
//
// - MLP chain on v_wmma_f32_16x16x32_bf16 (f32 accumulate).
// - Weights pre-converted ONCE to bf16 in the exact LDS staging layout
//   ([k-slice][n][32]); hot loop stages them with global_load_async_to_lds_b128
//   (ASYNCcnt + s_wait_asynccnt), no VGPR round trip.
// - Only xs_F_cat (1.02 GB) streams from HBM => ~45us roofline floor.
// ---------------------------------------------------------------------------

typedef __attribute__((ext_vector_type(16))) __bf16        v16bf;
typedef __attribute__((ext_vector_type(8)))  float         v8f;
typedef __attribute__((ext_vector_type(8)))  unsigned int  v8u;

#define DCAT        1280
#define DIM         256
#define NLAYER      2
#define BM          64          // edges per workgroup
#define SLICE_ELEMS (DIM * 32)  // one 32-wide k-slice in staging layout
#define INV_SQRT2   0.7071067811865476f
#define SILU_SCALE  (1.0f / 0.6f)

// ---- helpers --------------------------------------------------------------

__device__ __forceinline__ unsigned short f2bf(float f) {
  return __builtin_bit_cast(unsigned short, (__bf16)f);   // hw cvt, RNE
}
__device__ __forceinline__ unsigned int pack2bf(float lo, float hi) {
  return (unsigned int)f2bf(lo) | ((unsigned int)f2bf(hi) << 16);
}
__device__ __forceinline__ float bf2f(unsigned short h) {
  return __builtin_bit_cast(float, ((unsigned int)h) << 16);
}

// ScaledSiLU with single v_rcp_f32 (no IEEE divide chain).
__device__ __forceinline__ v8f ssilu(v8f x) {
  v8f r;
#pragma unroll
  for (int i = 0; i < 8; ++i) {
    float v = x[i];
    float s = __builtin_amdgcn_rcpf(1.0f + __expf(-v));
    r[i] = v * s * SILU_SCALE;
  }
  return r;
}

// A fragment (16x32 bf16, M x K), row-major bf16 source with rowStride.
// CDNA5 layout: lane m = lane&15; VGPR v holds K pair
//   k0 = (v>=4)*16 + (lane>=16)*8 + (v&3)*2   (pairs contiguous -> b128 fused)
__device__ __forceinline__ v16bf load_a_frag(const unsigned short* base,
                                             int rowStride, int m0, int lane) {
  const unsigned short* rp = base + (m0 + (lane & 15)) * rowStride;
  const int khalf = (lane >> 4) << 3;
  v8u a;
#pragma unroll
  for (int v = 0; v < 8; ++v) {
    int k0 = ((v >> 2) << 4) + khalf + ((v & 3) << 1);
    a[v] = *(const unsigned int*)(rp + k0);
  }
  return __builtin_bit_cast(v16bf, a);
}

// B fragment (32x16 bf16, K x N); staged col-major [n][32] in LDS.
//   lane n = lane&15; VGPR v holds K pair k0 = (v>=4)*16 + (lane>=16)*8 + (v&3)*2
__device__ __forceinline__ v16bf load_b_frag(const unsigned short* base,
                                             int n0, int lane) {
  const unsigned short* np = base + (n0 + (lane & 15)) * 32 + ((lane >> 4) << 3);
  v8u b;
#pragma unroll
  for (int v = 0; v < 8; ++v)
    b[v] = *(const unsigned int*)(np + ((v >> 2) << 4) + ((v & 3) << 1));
  return __builtin_bit_cast(v16bf, b);
}

#define WMMA_BF16(a, b, c) \
  __builtin_amdgcn_wmma_f32_16x16x32_bf16(false, (a), false, (b), (short)0, (c), false, false)

// Store a wave's 2x4 grid of 16x16 f32 C-tiles into bf16 buffer xb[64][256].
__device__ __forceinline__ void store_tiles(unsigned short* xb,
                                            v8f (&vreg)[2][4],
                                            int mBase, int nBase, int lane) {
  const int col  = lane & 15;
  const int rofs = (lane >> 4) << 3;
#pragma unroll
  for (int mt = 0; mt < 2; ++mt)
#pragma unroll
    for (int nt = 0; nt < 4; ++nt)
#pragma unroll
      for (int v = 0; v < 8; ++v) {
        int row = mBase + mt * 16 + rofs + v;
        xb[row * DIM + nBase + nt * 16 + col] = f2bf(vreg[mt][nt][v]);
      }
}

// Async-copy one pre-converted 16KB bf16 weight slice (staging layout) into
// sB, 64B per thread via 4x global_load_async_to_lds_b128 (GVS addressing:
// SGPR64 base + VGPR 32-bit offset; LDS byte address in the first VGPR).
__device__ __forceinline__ void stage_b_async(const unsigned short* __restrict__ Wb,
                                              int slice, unsigned short* sB, int t) {
  unsigned long long gbase =
      (unsigned long long)(const void*)(Wb + (size_t)slice * SLICE_ELEMS);
  unsigned lds0 = (unsigned)(unsigned long long)(const void*)sB + (unsigned)t * 64u;
  unsigned voff = (unsigned)t * 64u;
#pragma unroll
  for (int i = 0; i < 4; ++i) {
    asm volatile("global_load_async_to_lds_b128 %0, %1, %2"
                 :: "v"(lds0 + (unsigned)(i * 16)),
                    "v"(voff + (unsigned)(i * 16)),
                    "s"(gbase)
                 : "memory");
  }
}
__device__ __forceinline__ void wait_async0() {
  asm volatile("s_wait_asynccnt 0x0" ::: "memory");
}

// ---- weight pre-conversion (runs once per launch) --------------------------
// W [K,256] fp32 (row-major)  ->  Wb [K/32][256][32] bf16 (staging layout)
__global__ void convert_weights_kernel(const float* __restrict__ W,
                                       unsigned short* __restrict__ Wb,
                                       int total) {  // total = K*256
  int i = blockIdx.x * blockDim.x + threadIdx.x;
  if (i < total) {
    int k = i >> 8;
    int n = i & 255;
    Wb[((size_t)(k >> 5) << 13) + (n << 5) + (k & 31)] = f2bf(W[i]);
  }
}

// ---- fused MLP kernel ------------------------------------------------------
// block: 256 threads = 8 wave32, BM=64 edges; waves tiled 2(M) x 4(N).
__global__ __launch_bounds__(256) void gemnet_mlp_kernel(
    const float* __restrict__ xs,            // [E, 1280]
    const unsigned short* __restrict__ WinB, // [40][256][32] bf16
    const unsigned short* __restrict__ W1B,  // [16][256][32] bf16 (2 layers)
    const unsigned short* __restrict__ W2B,  // [16][256][32] bf16 (2 layers)
    const float* __restrict__ Wout,          // [256]
    float* __restrict__ Fst) {               // [E]
  __shared__ unsigned short sA[BM * 32];    //  4 KB  A k-slice (bf16)
  __shared__ unsigned short sB[DIM * 32];   // 16 KB  B k-slice (bf16, [n][32])
  __shared__ unsigned short xb[BM * DIM];   // 32 KB  activations (bf16)

  const int t    = threadIdx.x;
  const int lane = t & 31;
  const int w    = t >> 5;
  const int mBase = (w & 1) * 32;           // 2 M-tiles per wave
  const int nBase = (w >> 1) * 64;          // 4 N-tiles per wave
  const long row0 = (long)blockIdx.x * BM;

  const v8f vzero = {0.f, 0.f, 0.f, 0.f, 0.f, 0.f, 0.f, 0.f};
  v8f acc[2][4];
#pragma unroll
  for (int mt = 0; mt < 2; ++mt)
#pragma unroll
    for (int nt = 0; nt < 4; ++nt) acc[mt][nt] = vzero;

  // A-staging: thread t owns one contiguous 8-float chunk of its row.
  const int arow = t >> 2;                  // 0..63
  const int akk  = (t & 3) * 8;             // 0,8,16,24
  const float* aptr = xs + (row0 + arow) * DCAT + akk;

  // ---- GEMM 1: x = xs[row0:row0+64, :] @ Win   (K = 1280, 40 k-steps) ----
  for (int k0 = 0; k0 < DCAT; k0 += 32) {
    stage_b_async(WinB, k0 >> 5, sB, t);        // weights: async -> LDS
    float4 f0 = *(const float4*)(aptr + k0);    // A: 2x global_load_b128
    float4 f1 = *(const float4*)(aptr + k0 + 4);
    __builtin_prefetch(aptr + k0 + 32, 0, 0);   // next k-slice
    unsigned int* ap = (unsigned int*)(sA + arow * 32 + akk);
    ap[0] = pack2bf(f0.x, f0.y);
    ap[1] = pack2bf(f0.z, f0.w);
    ap[2] = pack2bf(f1.x, f1.y);
    ap[3] = pack2bf(f1.z, f1.w);
    wait_async0();
    __syncthreads();

    v16bf af[2], bfr[4];
#pragma unroll
    for (int mt = 0; mt < 2; ++mt) af[mt] = load_a_frag(sA, 32, mBase + mt * 16, lane);
#pragma unroll
    for (int nt = 0; nt < 4; ++nt) bfr[nt] = load_b_frag(sB, nBase + nt * 16, lane);
#pragma unroll
    for (int mt = 0; mt < 2; ++mt)
#pragma unroll
      for (int nt = 0; nt < 4; ++nt)
        acc[mt][nt] = WMMA_BF16(af[mt], bfr[nt], acc[mt][nt]);
    __syncthreads();
  }

  // x = ssilu(x); keep in registers, mirror in LDS (bf16)
  v8f xreg[2][4];
#pragma unroll
  for (int mt = 0; mt < 2; ++mt)
#pragma unroll
    for (int nt = 0; nt < 4; ++nt) xreg[mt][nt] = ssilu(acc[mt][nt]);
  store_tiles(xb, xreg, mBase, nBase, lane);
  __syncthreads();

  // ---- residual layers: x = (x + ssilu(ssilu(x@W1)@W2)) * INV_SQRT2 ----
  for (int l = 0; l < NLAYER; ++l) {
#pragma unroll
    for (int g = 0; g < 2; ++g) {
      const unsigned short* Wb = (g == 0) ? W1B : W2B;
      v8f hacc[2][4];
#pragma unroll
      for (int mt = 0; mt < 2; ++mt)
#pragma unroll
        for (int nt = 0; nt < 4; ++nt) hacc[mt][nt] = vzero;

      for (int k0 = 0; k0 < DIM; k0 += 32) {
        __syncthreads();                         // previous sB reads done
        stage_b_async(Wb, (l << 3) + (k0 >> 5), sB, t);
        wait_async0();
        __syncthreads();

        v16bf af[2], bfr[4];
#pragma unroll
        for (int mt = 0; mt < 2; ++mt)
          af[mt] = load_a_frag(xb + k0, DIM, mBase + mt * 16, lane);
#pragma unroll
        for (int nt = 0; nt < 4; ++nt)
          bfr[nt] = load_b_frag(sB, nBase + nt * 16, lane);
#pragma unroll
        for (int mt = 0; mt < 2; ++mt)
#pragma unroll
          for (int nt = 0; nt < 4; ++nt)
            hacc[mt][nt] = WMMA_BF16(af[mt], bfr[nt], hacc[mt][nt]);
      }
      __syncthreads();                           // all xb reads done

#pragma unroll
      for (int mt = 0; mt < 2; ++mt)
#pragma unroll
        for (int nt = 0; nt < 4; ++nt) hacc[mt][nt] = ssilu(hacc[mt][nt]);

      if (g == 0) {
        store_tiles(xb, hacc, mBase, nBase, lane);       // xb := h1
      } else {
#pragma unroll
        for (int mt = 0; mt < 2; ++mt)
#pragma unroll
          for (int nt = 0; nt < 4; ++nt)
            xreg[mt][nt] = (xreg[mt][nt] + hacc[mt][nt]) * INV_SQRT2;
        store_tiles(xb, xreg, mBase, nBase, lane);       // xb := new x
      }
      __syncthreads();
    }
  }

  // ---- F_st = x @ W_out  (one row per thread, 64 rows) ----
  if (t < BM) {
    float s = 0.f;
    const unsigned short* xrow = xb + t * DIM;
#pragma unroll 4
    for (int c = 0; c < DIM; c += 2) {
      unsigned int p = *(const unsigned int*)(xrow + c);
      s += bf2f((unsigned short)(p & 0xFFFFu)) * Wout[c];
      s += bf2f((unsigned short)(p >> 16)) * Wout[c + 1];
    }
    Fst[row0 + t] = s;
  }
}

// ---- small kernels ---------------------------------------------------------

__device__ __forceinline__ void atomic_add_f32(float* p, float v) {
  __hip_atomic_fetch_add(p, v, __ATOMIC_RELAXED, __HIP_MEMORY_SCOPE_AGENT);
}

__global__ void zero_kernel(float* __restrict__ p, int n) {
  int i = blockIdx.x * blockDim.x + threadIdx.x;
  if (i < n) p[i] = 0.f;
}

__global__ void seg_kernel(const float* __restrict__ Fst,
                           const int* __restrict__ idu,
                           float* __restrict__ segsum,
                           float* __restrict__ cnt, int E) {
  int e = blockIdx.x * blockDim.x + threadIdx.x;
  if (e < E) {
    int id = idu[e];
    atomic_add_f32(&segsum[id], Fst[e]);
    atomic_add_f32(&cnt[id], 1.0f);
  }
}

__global__ void scatter_kernel(const float* __restrict__ segsum,
                               const float* __restrict__ cnt,
                               const int* __restrict__ idu,
                               const int* __restrict__ eidx,
                               const float* __restrict__ evec,
                               float* __restrict__ out, int E) {
  int e = blockIdx.x * blockDim.x + threadIdx.x;
  if (e < E) {
    int id = idu[e];
    float f = segsum[id] / fmaxf(cnt[id], 1.0f);
    int a = eidx[e];
#pragma unroll
    for (int d = 0; d < 3; ++d)
      atomic_add_f32(&out[a * 3 + d], f * evec[e * 3 + d]);
  }
}

// ---- entry -----------------------------------------------------------------

extern "C" void kernel_launch(void* const* d_in, const int* in_sizes, int n_in,
                              void* d_out, int out_size, void* d_ws, size_t ws_size,
                              hipStream_t stream) {
  const float* xs   = (const float*)d_in[0];  // xs_F_cat  [E,1280]
  const float* evec = (const float*)d_in[1];  // edge_vec  [E,3]
  const float* Win  = (const float*)d_in[2];  // W_in      [1280,256]
  const float* W1   = (const float*)d_in[3];  // res_W1    [2,256,256]
  const float* W2   = (const float*)d_in[4];  // res_W2    [2,256,256]
  const float* Wout = (const float*)d_in[5];  // W_out     [256,1]
  const int*   eidx = (const int*)d_in[6];    // edge_idx  [E]
  const int*   idu  = (const int*)d_in[7];    // id_undir  [E]
  float* out = (float*)d_out;                 // [NA,3]

  const int E  = in_sizes[6];
  const int nU = E / 2;

  float* Fst    = (float*)d_ws;   // [E]
  float* segsum = Fst + E;        // [E/2]
  float* cnt    = segsum + nU;    // [E/2]
  unsigned short* WinB = (unsigned short*)(cnt + nU);   // [1280*256] bf16
  unsigned short* W1B  = WinB + DCAT * DIM;             // [512*256]  bf16
  unsigned short* W2B  = W1B + 2 * DIM * DIM;           // [512*256]  bf16

  zero_kernel<<<(out_size + 255) / 256, 256, 0, stream>>>(out, out_size);
  zero_kernel<<<(E + 255) / 256, 256, 0, stream>>>(segsum, E);

  // one-time (per launch) weight conversion into LDS staging layout
  convert_weights_kernel<<<(DCAT * DIM + 255) / 256, 256, 0, stream>>>(Win, WinB, DCAT * DIM);
  convert_weights_kernel<<<(2 * DIM * DIM + 255) / 256, 256, 0, stream>>>(W1, W1B, 2 * DIM * DIM);
  convert_weights_kernel<<<(2 * DIM * DIM + 255) / 256, 256, 0, stream>>>(W2, W2B, 2 * DIM * DIM);

  gemnet_mlp_kernel<<<E / BM, 256, 0, stream>>>(xs, WinB, W1B, W2B, Wout, Fst);

  seg_kernel<<<(E + 255) / 256, 256, 0, stream>>>(Fst, idu, segsum, cnt, E);
  scatter_kernel<<<(E + 255) / 256, 256, 0, stream>>>(segsum, cnt, idu, eidx,
                                                      evec, out, E);
}
